// VectorizedWindowDynamics_17600775979273
// MI455X (gfx1250) — matrix operand order repaired and checked by reading, not compiled, not run on hardware
//
#include <hip/hip_runtime.h>
#include <hip/hip_bf16.h>

// ---------------- problem constants (match reference) ----------------
#define NB    2048
#define NW    6
#define ND    1024
#define NH    8
#define NR    64
#define NDH   128
#define NROW  (NB*NW)          // 12288 rows of length 1024
#define BWD   (NROW*ND)        // 12582912
#define DT_C     0.09f
#define CPL_C    0.01f
#define TOL_C    0.05f
#define THIGH_C  0.22f
#define NSTEP 16

// ---------------- WMMA types ----------------
typedef __attribute__((ext_vector_type(16))) __bf16 v16bf;
typedef __attribute__((ext_vector_type(8)))  float  v8f;

union ABPack { v16bf v; uint4 q[2]; unsigned short u[16]; };

// ---------------- helpers ----------------
__device__ __forceinline__ unsigned short f2bf(float f){
  unsigned int u = __float_as_uint(f);
  u += 0x7FFFu + ((u >> 16) & 1u);          // round-to-nearest-even
  return (unsigned short)(u >> 16);
}
__device__ __forceinline__ float bf2f(unsigned short h){
  return __uint_as_float(((unsigned int)h) << 16);
}
__device__ __forceinline__ unsigned int pcg(unsigned int v){
  v = v*747796405u + 2891336453u;
  unsigned int w = ((v >> ((v >> 28) + 4u)) ^ v) * 277803737u;
  return (w >> 22) ^ w;
}
__device__ __forceinline__ float u01(unsigned int h){
  return fmaxf((float)(h >> 8) * (1.0f/16777216.0f), 1e-7f);
}

// =====================================================================
// P1: Mt[h][n][k] = -(sum_r U[h][k][r] * V[h][r][n])  as bf16
// =====================================================================
__global__ __launch_bounds__(256) void k_prep(const float* __restrict__ U,
                                              const float* __restrict__ V,
                                              unsigned short* __restrict__ Mt){
  int idx = blockIdx.x*256 + threadIdx.x;    // 8*128*128 = 131072 total
  int h = idx >> 14;
  int n = (idx >> 7) & 127;
  int k = idx & 127;
  float acc = 0.f;
  const float* u = U + (size_t)(h*NDH + k)*NR;
  const float* v = V + (size_t)h*NR*NDH + n;
  #pragma unroll 8
  for (int r = 0; r < NR; r++) acc += u[r] * v[r*NDH];
  Mt[idx] = f2bf(-acc);
}

// =====================================================================
// K0: per-call deterministic init of step-scalar accumulators
// =====================================================================
__global__ void k_init(float* Tsum, int* allf, int* anyf, int* done){
  int t = threadIdx.x;
  if (t < NSTEP){ Tsum[t] = 0.f; allf[t] = 1; anyf[t] = 0; }
  if (t <= NSTEP) done[t] = 0;
}

// =====================================================================
// K1: one Euler step.  Block = 16 rows. Wave w = head w.
//     Sn = clamp_norm(S + DT*( -x_h M_h + CPL*(mean_h - x_h) + signal ))
//     also writes invn[row] = 1/(clipped_norm + 1e-12) for tension kernel
// =====================================================================
__global__ __launch_bounds__(256) void k_dyn(const float* __restrict__ Sin,
                                             const float* __restrict__ Sig,
                                             const unsigned short* __restrict__ Mt,
                                             float* __restrict__ Sout,
                                             float* __restrict__ invn){
  __shared__ unsigned short Xb[16*ND];   // 32 KB bf16 copy of the 16-row tile
  __shared__ float Mh[16*NDH];           //  8 KB per-row cross-head mean
  __shared__ float Pp[16*NH];            // per-(row,head) partial norms

  const int t = threadIdx.x;
  const int rowbase = blockIdx.x * 16;
  const int wv  = t >> 5;                // wave id == head id
  const int lane = t & 31;
  const int h   = wv;
  const int r16 = lane & 15;
  const int khi = lane >> 4;

  __builtin_prefetch(Mt + (size_t)h*(NDH*NDH), 0, 0);

  // ---- phase 1: fp32 tile -> bf16 LDS (WMMA A staging) ----
  {
    int r = t >> 4, c0 = (t & 15) * 64;
    const float* src = Sin + (size_t)(rowbase + r)*ND + c0;
    unsigned short* dst = &Xb[r*ND + c0];
    #pragma unroll 4
    for (int j = 0; j < 64; j += 4){
      float4 f = *reinterpret_cast<const float4*>(src + j);
      dst[j+0] = f2bf(f.x); dst[j+1] = f2bf(f.y);
      dst[j+2] = f2bf(f.z); dst[j+3] = f2bf(f.w);
    }
  }
  __syncthreads();

  // ---- phase 2: per-row cross-head mean table Mh[r][e] ----
  #pragma unroll
  for (int i = 0; i < 8; i++){
    int idx = t*8 + i;                   // 2048 entries
    int r = idx >> 7, e = idx & 127;
    float s = 0.f;
    #pragma unroll
    for (int hh = 0; hh < NH; hh++) s += bf2f(Xb[r*ND + hh*NDH + e]);
    Mh[idx] = s * 0.125f;
  }

  // ---- phase 3: WMMA  Y(16x128, head h) = X_h(16x128) * M_h^T layout ----
  // A 16x32 bf16 per-lane layout: lane<16 -> K 0..7 & 16..23 ; lane>=16 -> K 8..15 & 24..31
  ABPack a[4];
  #pragma unroll
  for (int kc = 0; kc < 4; kc++){
    const unsigned short* p = &Xb[r16*ND + h*NDH + kc*32 + khi*8];
    a[kc].q[0] = *reinterpret_cast<const uint4*>(p);        // K +0..7
    a[kc].q[1] = *reinterpret_cast<const uint4*>(p + 16);   // K +16..23
  }
  v8f acc8[8];
  #pragma unroll
  for (int nt = 0; nt < 8; nt++){
    v8f c = (v8f)(0.0f);
    #pragma unroll
    for (int kc = 0; kc < 4; kc++){
      // B 32x16 bf16 per-lane layout: lane n<16 -> (N=n, K 0..7 & 16..23) ; lane>=16 -> (N=n-16, K 8..15 & 24..31)
      ABPack bp;
      const unsigned short* q =
          Mt + ((size_t)(h*NDH + nt*16 + r16))*NDH + kc*32 + khi*8;
      bp.q[0] = *reinterpret_cast<const uint4*>(q);         // K +0..7
      bp.q[1] = *reinterpret_cast<const uint4*>(q + 16);    // K +16..23
      c = __builtin_amdgcn_wmma_f32_16x16x32_bf16(
              false, a[kc].v, false, bp.v, (short)0, c, false, false);
    }
    acc8[nt] = c;
  }
  __syncthreads();   // Mh visible to all; WMMA accumulators live in VGPRs

  // ---- phase 4a: elementwise drift in the C-matrix layout + row partials ----
  // C layout: vgpr i, lanes0-15 -> (M=i, N=lane); lanes16-31 -> (M=8+i, N=lane-16)
  float s2[8];
  #pragma unroll
  for (int i = 0; i < 8; i++) s2[i] = 0.f;
  #pragma unroll
  for (int nt = 0; nt < 8; nt++){
    int e   = nt*16 + r16;               // position within head
    int col = h*NDH + e;                 // position within full D
    #pragma unroll
    for (int i = 0; i < 8; i++){
      int rloc = khi*8 + i;
      size_t g = (size_t)(rowbase + rloc)*ND + col;
      float x  = Sin[g];
      float sn = x + DT_C*(acc8[nt][i] + CPL_C*(Mh[rloc*NDH + e] - x) + Sig[g]);
      acc8[nt][i] = sn;
      s2[i] += sn*sn;
    }
  }
  #pragma unroll
  for (int i = 0; i < 8; i++){           // reduce across the 16 lanes of a row
    s2[i] += __shfl_xor(s2[i], 8, 32);
    s2[i] += __shfl_xor(s2[i], 4, 32);
    s2[i] += __shfl_xor(s2[i], 2, 32);
    s2[i] += __shfl_xor(s2[i], 1, 32);
  }
  if (r16 == 0){
    #pragma unroll
    for (int i = 0; i < 8; i++) Pp[(khi*8 + i)*NH + h] = s2[i];
  }
  __syncthreads();

  // ---- phase 4b: row norms across heads, clamp, write out ----
  float scl[8];
  #pragma unroll
  for (int i = 0; i < 8; i++){
    int rloc = khi*8 + i;
    float n2 = 0.f;
    #pragma unroll
    for (int hh = 0; hh < NH; hh++) n2 += Pp[rloc*NH + hh];
    float nr   = sqrtf(n2);
    float clip = fminf(fmaxf(nr, 1e-3f), 12.0f);
    scl[i] = clip / fmaxf(nr, 1e-8f);
    if (h == 0 && r16 == 0) invn[rowbase + rloc] = 1.0f/(clip + 1e-12f);
  }
  #pragma unroll
  for (int nt = 0; nt < 8; nt++){
    int col = h*NDH + nt*16 + r16;
    #pragma unroll
    for (int i = 0; i < 8; i++){
      size_t g = (size_t)(rowbase + khi*8 + i)*ND + col;
      Sout[g] = acc8[nt][i] * scl[i];
    }
  }
}

// =====================================================================
// K2: tension per batch element.  T_b = 1 - ||sum_w Sn_bw/||Sn_bw||||^2 / W^2
//     folds into step-scalar accumulators with atomics.
// =====================================================================
__global__ __launch_bounds__(256) void k_tension(const float* __restrict__ Sn,
                                                 const float* __restrict__ invn,
                                                 float* Tsum, int* allf, int* anyf){
  __shared__ float sm[8];
  int b = blockIdx.x, t = threadIdx.x;
  float inv[NW];
  #pragma unroll
  for (int w = 0; w < NW; w++) inv[w] = invn[b*NW + w];
  int d = t*4;
  float gx=0.f, gy=0.f, gz=0.f, gw=0.f;
  #pragma unroll
  for (int w = 0; w < NW; w++){
    float4 s = *reinterpret_cast<const float4*>(Sn + (size_t)(b*NW + w)*ND + d);
    gx += s.x*inv[w]; gy += s.y*inv[w]; gz += s.z*inv[w]; gw += s.w*inv[w];
  }
  float v = gx*gx + gy*gy + gz*gz + gw*gw;
  v += __shfl_xor(v,16,32); v += __shfl_xor(v,8,32); v += __shfl_xor(v,4,32);
  v += __shfl_xor(v,2,32);  v += __shfl_xor(v,1,32);
  if ((t & 31) == 0) sm[t >> 5] = v;
  __syncthreads();
  float tot = 0.f;
  #pragma unroll
  for (int i = 0; i < 8; i++) tot += sm[i];
  float T = 1.0f - tot * (1.0f/(float)(NW*NW));
  if (t == 0){
    atomicAdd(Tsum, T);
    if (!(T < TOL_C))  atomicAnd(allf, 0);   // all(T<TOL) flag
    if (T > THIGH_C)   atomicOr(anyf, 1);    // any(T>THIGH) flag
  }
}

// =====================================================================
// K3: apply done/noise logic, in-place on Sn; records curve[k]; advances done
// =====================================================================
__global__ __launch_bounds__(256) void k_apply(const float* __restrict__ Sc,
                                               float* __restrict__ Sn,
                                               const float* __restrict__ Tsum,
                                               const int* __restrict__ allf,
                                               const int* __restrict__ anyf,
                                               const int* __restrict__ donek,
                                               int* donek1, float* curvek, int step){
  __shared__ float sm[8];
  int row = blockIdx.x, t = threadIdx.x;
  int done = *donek, alla = *allf, anya = *anyf;
  int new_done = done | alla;
  int apply = (!new_done) && anya;

  size_t base = (size_t)row*ND + t*4;
  float4 sn = *reinterpret_cast<const float4*>(Sn + base);

  // hash-based gaussian noise kick (approximates jax fold_in stream)
  unsigned int seed = 0x9E3779B9u * (unsigned)(step + 1);
  unsigned int i0 = (unsigned)row*1024u + (unsigned)(t*4);
  unsigned int h0 = pcg(i0 ^ seed),      h1 = pcg((i0+1u) ^ seed);
  unsigned int h2 = pcg((i0+2u) ^ seed), h3 = pcg((i0+3u) ^ seed);
  float r0 = sqrtf(-2.f*__logf(u01(h0))), a0 = 6.2831853f*u01(h1);
  float r1 = sqrtf(-2.f*__logf(u01(h2))), a1 = 6.2831853f*u01(h3);
  float4 sz;
  sz.x = sn.x + 0.01f*(r0*__cosf(a0));
  sz.y = sn.y + 0.01f*(r0*__sinf(a0));
  sz.z = sn.z + 0.01f*(r1*__cosf(a1));
  sz.w = sn.w + 0.01f*(r1*__sinf(a1));

  float v = sz.x*sz.x + sz.y*sz.y + sz.z*sz.z + sz.w*sz.w;
  v += __shfl_xor(v,16,32); v += __shfl_xor(v,8,32); v += __shfl_xor(v,4,32);
  v += __shfl_xor(v,2,32);  v += __shfl_xor(v,1,32);
  if ((t & 31) == 0) sm[t >> 5] = v;
  __syncthreads();
  float tot = 0.f;
  #pragma unroll
  for (int i = 0; i < 8; i++) tot += sm[i];
  float norm = sqrtf(tot);

  float4 outv;
  if (done){
    outv = *reinterpret_cast<const float4*>(Sc + base);     // frozen
  } else if (apply){
    float is = 1.0f/(norm + 1e-8f);
    outv = make_float4(sz.x*is, sz.y*is, sz.z*is, sz.w*is); // noise + renorm
  } else {
    outv = sn;
  }
  *reinterpret_cast<float4*>(Sn + base) = outv;

  if (row == 0 && t == 0){
    *donek1 = new_done;
    *curvek = done ? __uint_as_float(0x7FC00000u) : (*Tsum) * (1.0f/(float)NB);
  }
}

// =====================================================================
extern "C" void kernel_launch(void* const* d_in, const int* in_sizes, int n_in,
                              void* d_out, int out_size, void* d_ws, size_t ws_size,
                              hipStream_t stream){
  (void)in_sizes; (void)n_in; (void)out_size; (void)ws_size;
  const float* S   = (const float*)d_in[0];
  const float* Sig = (const float*)d_in[1];
  const float* U   = (const float*)d_in[2];
  const float* V   = (const float*)d_in[3];

  char* ws = (char*)d_ws;
  float* P0 = (float*)ws;                    ws += (size_t)BWD*sizeof(float);
  float* P1 = (float*)ws;                    ws += (size_t)BWD*sizeof(float);
  unsigned short* Mt = (unsigned short*)ws;  ws += (size_t)NH*NDH*NDH*sizeof(unsigned short);
  float* invn = (float*)ws;                  ws += (size_t)NROW*sizeof(float);
  float* Tsum = (float*)ws;                  ws += NSTEP*sizeof(float);
  int*   allf = (int*)ws;                    ws += NSTEP*sizeof(int);
  int*   anyf = (int*)ws;                    ws += NSTEP*sizeof(int);
  int*   done = (int*)ws;                    ws += (NSTEP+1)*sizeof(int);

  float* dS    = (float*)d_out;              // final S, BWD floats
  float* curve = dS + BWD;                   // 16 floats

  hipLaunchKernelGGL(k_init, dim3(1),   dim3(64),  0, stream, Tsum, allf, anyf, done);
  hipLaunchKernelGGL(k_prep, dim3(512), dim3(256), 0, stream, U, V, Mt);

  const float* cur = S;
  for (int k = 0; k < NSTEP; k++){
    float* sn = (k == NSTEP-1) ? dS : ((k & 1) ? P1 : P0);
    hipLaunchKernelGGL(k_dyn,     dim3(NROW/16), dim3(256), 0, stream,
                       cur, Sig, Mt, sn, invn);
    hipLaunchKernelGGL(k_tension, dim3(NB),      dim3(256), 0, stream,
                       sn, invn, Tsum+k, allf+k, anyf+k);
    hipLaunchKernelGGL(k_apply,   dim3(NROW),    dim3(256), 0, stream,
                       cur, sn, Tsum+k, allf+k, anyf+k, done+k, done+k+1, curve+k, k);
    cur = sn;
  }
}
